// Attn_14723147891087
// MI455X (gfx1250) — matrix operand order repaired and checked by reading, not compiled, not run on hardware
//
#include <hip/hip_runtime.h>

#define HDIM 128
#define SDIM 4096
#define BDIM 128
#define LDSW 136   // padded LDS row stride in bf16 elems (272B -> 4-bank rotate/row)

typedef __attribute__((ext_vector_type(16))) __bf16 bf16x16;
typedef __attribute__((ext_vector_type(8)))  float  f32x8;
typedef __attribute__((ext_vector_type(4)))  int    v4i;
typedef __attribute__((address_space(1)))    v4i    gv4i;   // global int4
typedef __attribute__((address_space(3)))    v4i    lv4i;   // LDS int4

union Frag {
    bf16x16 v;
    __bf16  h[16];
    unsigned short u[16];
    uint4   q[2];
};

#ifndef __has_builtin
#define __has_builtin(x) 0
#endif
#if __has_builtin(__builtin_amdgcn_global_load_async_to_lds_b128)
#define HAVE_ASYNC_LDS 1
#else
#define HAVE_ASYNC_LDS 0
#endif

__device__ __forceinline__ void wait_asynccnt0() {
#if __has_builtin(__builtin_amdgcn_s_wait_asynccnt)
    __builtin_amdgcn_s_wait_asynccnt(0);
#else
    asm volatile("s_wait_asynccnt 0x0" ::: "memory");
#endif
}

__device__ __forceinline__ float fast_tanh(float z) {
    // tanh(z) = 1 - 2/(exp(2z)+1); saturates correctly at +/-inf
    return 1.0f - 2.0f / (__expf(2.0f * z) + 1.0f);
}

// Kernel 0: one-time W f32 -> bf16 conversion (64 blocks x 256 threads).
__global__ __launch_bounds__(256) void
convert_W(const float* __restrict__ W, __bf16* __restrict__ Wbf)
{
    const int i = blockIdx.x * 256 + threadIdx.x;   // 16384 elements
    Wbf[i] = (__bf16)W[i];
}

// Kernel 1: fused bf16-WMMA GEMM (y = x * W^T), +bias, tanh, dot with v -> scores[M]
__global__ __launch_bounds__(256) void
gemm_tanh_score(const float* __restrict__ x, const __bf16* __restrict__ Wbf,
                const float* __restrict__ bias, const float* __restrict__ v,
                float* __restrict__ scores)
{
    __shared__ unsigned short Wl[HDIM * LDSW];

    const int tid = threadIdx.x;

    // Stage pre-converted bf16 W (128x128) into padded LDS, 16B chunks.
    // 2048 chunks of 8 bf16; 8 chunks per thread.
#if HAVE_ASYNC_LDS
    for (int c = tid; c < 2048; c += 256) {
        const int r = c >> 4, cc = (c & 15) << 3;
        __builtin_amdgcn_global_load_async_to_lds_b128(
            (gv4i*)(Wbf + r * HDIM + cc),
            (lv4i*)&Wl[r * LDSW + cc],
            0, 0);
    }
    wait_asynccnt0();
#else
    for (int c = tid; c < 2048; c += 256) {
        const int r = c >> 4, cc = (c & 15) << 3;
        *(uint4*)&Wl[r * LDSW + cc] = *(const uint4*)(Wbf + r * HDIM + cc);
    }
#endif
    __syncthreads();

    const int lane = tid & 31;
    const int wave = tid >> 5;
    const int g    = lane >> 4;   // half-wave group
    const int ln   = lane & 15;

    const int rowBase = blockIdx.x * 128 + wave * 16;
    const float* xr = x + (size_t)(rowBase + ln) * HDIM;

    // --- Build A fragments (16x32 bf16 per K-step, documented CDNA5 layout):
    // lane group g, halves 0..7  -> K = k0..k0+7      (k0 = 32*kk + 8*g)
    //               halves 8..15 -> K = k0+16..k0+23
    // Native __bf16 casts -> compiler emits packed v_cvt for f32->bf16.
    Frag a[4];
#pragma unroll
    for (int kk = 0; kk < 4; ++kk) {
        const int k0 = kk * 32 + g * 8;
        float4 f0 = *(const float4*)(xr + k0);
        float4 f1 = *(const float4*)(xr + k0 + 4);
        float4 f2 = *(const float4*)(xr + k0 + 16);
        float4 f3 = *(const float4*)(xr + k0 + 20);
        a[kk].h[0]  = (__bf16)f0.x; a[kk].h[1]  = (__bf16)f0.y;
        a[kk].h[2]  = (__bf16)f0.z; a[kk].h[3]  = (__bf16)f0.w;
        a[kk].h[4]  = (__bf16)f1.x; a[kk].h[5]  = (__bf16)f1.y;
        a[kk].h[6]  = (__bf16)f1.z; a[kk].h[7]  = (__bf16)f1.w;
        a[kk].h[8]  = (__bf16)f2.x; a[kk].h[9]  = (__bf16)f2.y;
        a[kk].h[10] = (__bf16)f2.z; a[kk].h[11] = (__bf16)f2.w;
        a[kk].h[12] = (__bf16)f3.x; a[kk].h[13] = (__bf16)f3.y;
        a[kk].h[14] = (__bf16)f3.z; a[kk].h[15] = (__bf16)f3.w;
    }

    const f32x8 vzero = {0.f, 0.f, 0.f, 0.f, 0.f, 0.f, 0.f, 0.f};
    f32x8 acc[8];
#pragma unroll
    for (int t = 0; t < 8; ++t) acc[t] = vzero;

    // --- Main WMMA loop: 8 N-tiles x 4 K-steps = 32 v_wmma_f32_16x16x32_bf16.
    // B layout (32x16 bf16): lane = N (0..15) in each half-wave; group g halves
    // 0..7 hold K=8g..8g+7, halves 8..15 hold K=16+8g..16+8g+7 of the K-step.
#pragma unroll
    for (int t = 0; t < 8; ++t) {
        const unsigned short* wrowBase = &Wl[(t * 16 + ln) * LDSW];
#pragma unroll
        for (int kk = 0; kk < 4; ++kk) {
            const unsigned short* wrow = wrowBase + kk * 32 + g * 8;
            Frag bfr;
            bfr.q[0] = *(const uint4*)(wrow);       // K = base .. base+7
            bfr.q[1] = *(const uint4*)(wrow + 16);  // K = base+16 .. base+23
            acc[t] = __builtin_amdgcn_wmma_f32_16x16x32_bf16(
                false, a[kk].v, false, bfr.v, (short)0, acc[t], false, false);
        }
    }

    // --- Epilogue: +bias, tanh, dot with v, reduce across N (lanes), store scores.
    // D layout: acc[t] element r -> (m = r + 8*g, n = t*16 + ln).
    float partial[8] = {0.f, 0.f, 0.f, 0.f, 0.f, 0.f, 0.f, 0.f};
#pragma unroll
    for (int t = 0; t < 8; ++t) {
        const int n  = t * 16 + ln;
        const float bn = bias[n];
        const float vn = v[n];
#pragma unroll
        for (int r = 0; r < 8; ++r) {
            partial[r] += fast_tanh(acc[t][r] + bn) * vn;
        }
    }
#pragma unroll
    for (int r = 0; r < 8; ++r) {
        float s = partial[r];
        s += __shfl_xor(s, 1, 32);
        s += __shfl_xor(s, 2, 32);
        s += __shfl_xor(s, 4, 32);
        s += __shfl_xor(s, 8, 32);
        if (ln == 0) {
            scores[rowBase + r + 8 * g] = s;
        }
    }
}

// Kernel 2: softmax along S=4096 for each of B=128 rows.
__global__ __launch_bounds__(256) void
softmax_rows(const float* __restrict__ scores, float* __restrict__ out)
{
    __shared__ float red[256];
    const int b = blockIdx.x;
    const int tid = threadIdx.x;
    const float* row = scores + (size_t)b * SDIM;

    float vals[16];
    float lmax = -3.402823466e38f;
#pragma unroll
    for (int i = 0; i < 16; ++i) {
        vals[i] = row[tid + 256 * i];
        lmax = fmaxf(lmax, vals[i]);
    }
    red[tid] = lmax;
    __syncthreads();
    for (int s = 128; s > 0; s >>= 1) {
        if (tid < s) red[tid] = fmaxf(red[tid], red[tid + s]);
        __syncthreads();
    }
    const float m = red[0];
    __syncthreads();

    float lsum = 0.f;
#pragma unroll
    for (int i = 0; i < 16; ++i) {
        vals[i] = __expf(vals[i] - m);
        lsum += vals[i];
    }
    red[tid] = lsum;
    __syncthreads();
    for (int s = 128; s > 0; s >>= 1) {
        if (tid < s) red[tid] += red[tid + s];
        __syncthreads();
    }
    const float inv = 1.0f / red[0];

    float* orow = out + (size_t)b * SDIM;
#pragma unroll
    for (int i = 0; i < 16; ++i) {
        orow[tid + 256 * i] = vals[i] * inv;
    }
}

extern "C" void kernel_launch(void* const* d_in, const int* in_sizes, int n_in,
                              void* d_out, int out_size, void* d_ws, size_t ws_size,
                              hipStream_t stream)
{
    const float* x = (const float*)d_in[0];   // [B,S,H]
    const float* W = (const float*)d_in[1];   // [H,H]
    const float* b = (const float*)d_in[2];   // [H]
    const float* v = (const float*)d_in[3];   // [H]
    float* out     = (float*)d_out;           // [B,S]

    // Workspace layout: [0,64KB) bf16 W copy; [64KB, 64KB+2MiB) f32 scores.
    __bf16* Wbf   = (__bf16*)d_ws;
    float* scores = (float*)((char*)d_ws + 65536);

    dim3 blk(256);
    convert_W<<<dim3(HDIM * HDIM / 256), blk, 0, stream>>>(W, Wbf);

    const int M = BDIM * SDIM;                // 524288 rows
    gemm_tanh_score<<<dim3(M / 128), blk, 0, stream>>>(x, Wbf, b, v, scores);

    softmax_rows<<<dim3(BDIM), blk, 0, stream>>>(scores, out);
}
